// RNN_63144609186401
// MI455X (gfx1250) — compile-verified, matrix operand order checked by reading, not run on hardware
//
#include <hip/hip_runtime.h>

typedef __attribute__((ext_vector_type(16))) _Float16 v16h;
typedef __attribute__((ext_vector_type(8)))  _Float16 v8h;
typedef __attribute__((ext_vector_type(8)))  float    v8f;

#define T_STEPS 1000
#define ISZ 64      // input size (== padded K for x GEMM)
#define HSZ 50      // hidden size
#define G4  200     // 4*H gate columns
#define NT2 16      // padded n-tiles of 16 -> 256 cols (13 real + 3 zero pads)
#define NPAD 256
#define KPAD 64     // padded K for h GEMM (50 -> 64)

// Workspace layout (bytes):
//   [0, 32768)       : packed W_ih fragments, f16  [NT2][2 kblk][32 lane][16 half]
//   [32768, 65536)   : packed W_hh fragments, f16  (same shape)
//   [65536, 66560)   : fused bias (b_ih + b_hh), f32 [256] (zero-padded)
#define PACK_HALVES (NT2 * 2 * 32 * 16)   // 16384 per matrix

// ---------------------------------------------------------------------------
// Pack W_ih / W_hh into WMMA B-fragment order (B[k][n], 16-bit layout:
// lane = n%16 + 16*((k>>3)&1), half slot = (k&7) + 8*(k>>4) within a 32-K blk)
// so the LSTM kernel can load each lane's v16h with two coalesced b128 loads.
// Pad tiles (n >= 200) and pad K rows are zero-filled. Also fuses the biases.
// ---------------------------------------------------------------------------
__global__ void lstm_pack_kernel(const float* __restrict__ W_ih,
                                 const float* __restrict__ W_hh,
                                 const float* __restrict__ b_ih,
                                 const float* __restrict__ b_hh,
                                 _Float16* __restrict__ Bih,
                                 _Float16* __restrict__ Bhh,
                                 float* __restrict__ biasw) {
  int id = blockIdx.x * blockDim.x + threadIdx.x;
  if (id < 2 * PACK_HALVES) {
    int mat  = id / PACK_HALVES;
    int r    = id % PACK_HALVES;        // r = ((nt*2+kb)*32 + L)*16 + slot
    int slot = r & 15;
    int L    = (r >> 4) & 31;
    int kb   = (r >> 9) & 1;
    int nt   = r >> 10;                 // 0..15
    int n    = nt * 16 + (L & 15);
    int g    = L >> 4;
    int k    = 32 * kb + (slot & 7) + 8 * g + 16 * ((slot >> 3) & 1);
    float v  = 0.0f;
    if (mat == 0) {
      if (n < G4 && k < ISZ) v = W_ih[n * ISZ + k];   // B[k][n] = W_ih[n][k]
      Bih[r] = (_Float16)v;
    } else {
      if (n < G4 && k < HSZ) v = W_hh[n * HSZ + k];   // B[k][n] = W_hh[n][k]
      Bhh[r] = (_Float16)v;
    }
  } else if (id < 2 * PACK_HALVES + NPAD) {
    int n = id - 2 * PACK_HALVES;
    biasw[n] = (n < G4) ? (b_ih[n] + b_hh[n]) : 0.0f;
  }
}

// Branch-free transcendentals (v_exp_f32 / v_rcp_f32, no divergent slow paths)
__device__ __forceinline__ float fsigmoid(float v) {
  return __builtin_amdgcn_rcpf(1.0f + __expf(-v));          // exp->inf => rcp->0
}
__device__ __forceinline__ float ftanh(float v) {
  const float s = __expf(-2.0f * __builtin_fabsf(v));       // in (0, 1], no overflow
  const float t = (1.0f - s) * __builtin_amdgcn_rcpf(1.0f + s);
  return __builtin_copysignf(t, v);
}

// Gather a 16x32 f16 A-fragment from a row-major [16][KPAD] LDS tile.
// Lane L: row m = L&15, group g = L>>4; halves 0..7  <- K = 32*kb + 8g .. +7
//                                       halves 8..15 <- K = 32*kb + 16 + 8g ..
__device__ __forceinline__ v16h load_a_frag(const _Float16* base, int m, int g, int kb) {
  const v8h lo = *(const v8h*)(base + m * KPAD + 32 * kb + 8 * g);
  const v8h hi = *(const v8h*)(base + m * KPAD + 32 * kb + 16 + 8 * g);
  v16h r;
#pragma unroll
  for (int s = 0; s < 8; ++s) { r[s] = lo[s]; r[s + 8] = hi[s]; }
  return r;
}

// ---------------------------------------------------------------------------
// Fused LSTM: one block = one 16-row batch tile, 1000 sequential steps.
// Per step: gates[16][256] = bias + x_t[16][64]@Bih + h[16][64]@Bhh via
// v_wmma_f32_16x16x32_f16. Every wave owns exactly 2 n-tiles (uniform,
// straight-line, full-EXEC hot loop). Cell state lives in registers.
// x(t+1) is double-buffered through LDS; x(t+2) is prefetched.
// ---------------------------------------------------------------------------
__global__ __launch_bounds__(256)
void lstm_scan_kernel(const float* __restrict__ x,
                      const _Float16* __restrict__ Bih,
                      const _Float16* __restrict__ Bhh,
                      const float* __restrict__ biasw,
                      const float* __restrict__ W_out,
                      const float* __restrict__ b_out,
                      float* __restrict__ out) {
  __shared__ alignas(16) _Float16 xstage[2][16 * KPAD];  // f16 x tile, dbl-buffered
  __shared__ alignas(16) _Float16 hbuf[16 * KPAD];       // f16 h tile (K-padded)
  __shared__ float g_lds[16 * NPAD];                     // raw gates, row-major

  const int tid  = threadIdx.x;
  const int lane = tid & 31;
  const int wave = tid >> 5;
  const int b0   = blockIdx.x * 16;

  // --- init LDS: h = 0 (incl. K padding, stays 0 forever) ---
  for (int i = tid; i < 16 * KPAD; i += 256) hbuf[i] = (_Float16)0.0f;

  // --- stage x(0): thread -> (row m, 4 consecutive k) ---
  const int sm = tid >> 4;
  const int sq = (tid & 15) * 4;
  {
    const float4 x0 = *(const float4*)&x[((long)(b0 + sm) * T_STEPS + 0) * ISZ + sq];
    _Float16* d = &xstage[0][sm * KPAD + sq];
    d[0] = (_Float16)x0.x; d[1] = (_Float16)x0.y;
    d[2] = (_Float16)x0.z; d[3] = (_Float16)x0.w;
  }

  const int am = lane & 15, ag = lane >> 4;

  // --- preload constant B fragments + bias scalars into VGPRs ---
  v16h bihf[2][2], bhhf[2][2];
  float bsc[2];
#pragma unroll
  for (int i = 0; i < 2; ++i) {
    const int nt = wave + 8 * i;           // 0..15, always valid (padded)
    bsc[i] = biasw[nt * 16 + am];
#pragma unroll
    for (int kb = 0; kb < 2; ++kb) {
      const long vbase = ((long)(nt * 2 + kb) * 32 + lane) * 2;  // v8h units
      const v8h l0 = ((const v8h*)Bih)[vbase], h0 = ((const v8h*)Bih)[vbase + 1];
      const v8h l1 = ((const v8h*)Bhh)[vbase], h1 = ((const v8h*)Bhh)[vbase + 1];
      v16h t0, t1;
#pragma unroll
      for (int s = 0; s < 8; ++s) {
        t0[s] = l0[s]; t0[s + 8] = h0[s];
        t1[s] = l1[s]; t1[s + 8] = h1[s];
      }
      bihf[i][kb] = t0; bhhf[i][kb] = t1;
    }
  }

  // cell state in registers: thread owns gate elements tid, tid+256, tid+512, tid+768
  float creg[4] = {0.0f, 0.0f, 0.0f, 0.0f};

  __syncthreads();

  for (int t = 0; t < T_STEPS; ++t) {
    const int cur = t & 1, nxt = cur ^ 1;

    // prefetch x(t+1) into registers (h-independent; overlaps the WMMA chain)
    float4 xn = make_float4(0.0f, 0.0f, 0.0f, 0.0f);
    if (t + 1 < T_STEPS)
      xn = *(const float4*)&x[((long)(b0 + sm) * T_STEPS + (t + 1)) * ISZ + sq];
    if (t + 2 < T_STEPS)
      __builtin_prefetch(&x[((long)(b0 + sm) * T_STEPS + (t + 2)) * ISZ + sq], 0, 3);

    // A fragments (shared by both n-tiles this wave handles)
    const v16h ax0 = load_a_frag(xstage[cur], am, ag, 0);
    const v16h ax1 = load_a_frag(xstage[cur], am, ag, 1);
    const v16h ah0 = load_a_frag(hbuf, am, ag, 0);
    const v16h ah1 = load_a_frag(hbuf, am, ag, 1);

#pragma unroll
    for (int i = 0; i < 2; ++i) {
      const int nt = wave + 8 * i;
      v8f acc;
#pragma unroll
      for (int s = 0; s < 8; ++s) acc[s] = bsc[i];   // pure-VALU init
      acc = __builtin_amdgcn_wmma_f32_16x16x32_f16(false, ax0, false, bihf[i][0],
                                                   (short)0, acc, false, false);
      acc = __builtin_amdgcn_wmma_f32_16x16x32_f16(false, ax1, false, bihf[i][1],
                                                   (short)0, acc, false, false);
      acc = __builtin_amdgcn_wmma_f32_16x16x32_f16(false, ah0, false, bhhf[i][0],
                                                   (short)0, acc, false, false);
      acc = __builtin_amdgcn_wmma_f32_16x16x32_f16(false, ah1, false, bhhf[i][1],
                                                   (short)0, acc, false, false);
      // C layout: vgpr s -> row s + 8*(lane>>4), col nt*16 + (lane&15)
      const int col = nt * 16 + am;
      const int rb  = 8 * ag;
#pragma unroll
      for (int s = 0; s < 8; ++s) g_lds[(rb + s) * NPAD + col] = acc[s];
    }
    __syncthreads();

    // elementwise gate nonlinearity + state update (800 lanes of work);
    // mapping r = idx/50, j = idx%50 -> consecutive threads hit consecutive
    // LDS banks. Cell state stays in creg[]. Fully branch-free math.
#pragma unroll
    for (int u = 0; u < 4; ++u) {
      const int idx = tid + 256 * u;
      if (u < 3 || idx < 16 * HSZ) {
        const int r = idx / HSZ;
        const int j = idx - r * HSZ;
        const float* gr = &g_lds[r * NPAD];
        const float ig = gr[j], fg = gr[50 + j], gg = gr[100 + j], og = gr[150 + j];
        const float iv = fsigmoid(ig);
        const float fv = fsigmoid(fg);
        const float gv = ftanh(gg);
        const float ov = fsigmoid(og);
        const float cv = fv * creg[u] + iv * gv;
        creg[u] = cv;
        hbuf[r * KPAD + j] = (_Float16)(ov * ftanh(cv));
      }
    }

    // commit staged x(t+1) to the other LDS buffer
    {
      _Float16* d = &xstage[nxt][sm * KPAD + sq];
      d[0] = (_Float16)xn.x; d[1] = (_Float16)xn.y;
      d[2] = (_Float16)xn.z; d[3] = (_Float16)xn.w;
    }
    __syncthreads();
  }

  // final projection: out[b][e] = h_last . W_out[e] + b_out[e]
  if (tid < 32) {
    const int r = tid >> 1, e = tid & 1;
    float s = b_out[e];
#pragma unroll 10
    for (int j = 0; j < HSZ; ++j)
      s += (float)hbuf[r * KPAD + j] * W_out[e * HSZ + j];
    out[(b0 + r) * 2 + e] = s;
  }
}

extern "C" void kernel_launch(void* const* d_in, const int* in_sizes, int n_in,
                              void* d_out, int out_size, void* d_ws, size_t ws_size,
                              hipStream_t stream) {
  (void)in_sizes; (void)n_in; (void)out_size; (void)ws_size;
  const float* x     = (const float*)d_in[0];
  const float* W_ih  = (const float*)d_in[1];
  const float* W_hh  = (const float*)d_in[2];
  const float* b_ih  = (const float*)d_in[3];
  const float* b_hh  = (const float*)d_in[4];
  const float* W_out = (const float*)d_in[5];
  const float* b_out = (const float*)d_in[6];
  float* out = (float*)d_out;

  _Float16* Bih = (_Float16*)d_ws;
  _Float16* Bhh = Bih + PACK_HALVES;
  float* biasw  = (float*)((char*)d_ws + 2 * PACK_HALVES * sizeof(_Float16));

  const int total = 2 * PACK_HALVES + NPAD;
  lstm_pack_kernel<<<(total + 255) / 256, 256, 0, stream>>>(
      W_ih, W_hh, b_ih, b_hh, Bih, Bhh, biasw);
  lstm_scan_kernel<<<512 / 16, 256, 0, stream>>>(
      x, Bih, Bhh, biasw, W_out, b_out, out);
}
